// MultiHeadSelfAttention_67001489818305
// MI455X (gfx1250) — matrix operand order repaired and checked by reading, not compiled, not run on hardware
//
#include <hip/hip_runtime.h>
#include <hip/hip_fp16.h>

// ---------------- CDNA5 WMMA / TDM types ----------------
typedef __attribute__((ext_vector_type(16))) _Float16     v16h;
typedef __attribute__((ext_vector_type(8)))  float        v8f;
typedef __attribute__((ext_vector_type(4)))  unsigned int u32x4;
typedef __attribute__((ext_vector_type(4)))  int          i32x4;
typedef __attribute__((ext_vector_type(8)))  int          i32x8;

#define NBATCH 2
#define SEQ    2048
#define DIM    1024
#define NHEADS 16
#define HD     64
#define MROWS  (NBATCH * SEQ)     // 4096
#define MASKV  (-1.0e9f)

// GEMM tiling: block 128(M) x 64(N), K-step 64, 256 threads = 8 waves.
#define KSTEP  64
#define LDS_P  72                 // halves per LDS row: 64 data + 8 pad (16B)
#define ABUF_H (128 * LDS_P)      // halves per A buffer
#define WBUF_H (64 * LDS_P)       // halves per W buffer

union Frag16 {
    v16h  v;
    _Float16 h[16];
    uint4 q[2];
};

__device__ __forceinline__ v8f wmma_f16(const Frag16& a, const Frag16& b, v8f c) {
    return __builtin_amdgcn_wmma_f32_16x16x32_f16(
        false, a.v, false, b.v, (short)0, c, false, false);
}

// ======================================================================
// TDM 2D tile load: tile_rows x KSTEP f16 elements from a row-major
// [tensor_rows][DIM] f16 tensor into LDS with hardware row padding:
// 32 dwords (64 halves) data + 4 dwords pad -> LDS row stride = 72 halves.
// Descriptor packing per CDNA5 ISA section 8.3-8.5.
// ======================================================================
#if __has_builtin(__builtin_amdgcn_tensor_load_to_lds)
#define HAVE_TDM 1
__device__ __forceinline__ void tdm_load_tile(const _Float16* gsrc, unsigned lds_byte,
                                              int tensor_rows, int tile_rows)
{
    const unsigned long long ga = (unsigned long long)(const void*)gsrc;
    u32x4 g0;
    g0[0] = 1u;                                         // count=1, user desc
    g0[1] = lds_byte;                                   // lds_addr
    g0[2] = (unsigned)ga;                               // global_addr[31:0]
    g0[3] = (unsigned)((ga >> 32) & 0x01ffffffu)        // global_addr[56:32]
          | (2u << 30);                                 // type = 2 ("image")
    i32x8 g1;
    g1[0] = (1 << 16)                                   // data_size = 2 bytes
          | (1 << 20)                                   // pad_enable
          | (4 << 22)                                   // pad_interval: 32 dwords
          | (3 << 25);                                  // pad_amount: 4 dwords
    g1[1] = (DIM & 0xffff) << 16;                       // tensor_dim0 lo16 (abar=0)
    g1[2] = (DIM >> 16) | ((tensor_rows & 0xffff) << 16);   // td0 hi | td1 lo
    g1[3] = ((tensor_rows >> 16) & 0xffff) | (KSTEP << 16); // td1 hi | tile_dim0
    g1[4] = tile_rows & 0xffff;                         // tile_dim1 (tile_dim2=0)
    g1[5] = DIM;                                        // tensor_dim0_stride lo32
    g1[6] = 0;
    g1[7] = 0;
    const i32x4 z4 = {0, 0, 0, 0};
#if defined(__clang_major__) && (__clang_major__ >= 23)
    const i32x8 z8 = {0, 0, 0, 0, 0, 0, 0, 0};
    __builtin_amdgcn_tensor_load_to_lds(g0, g1, z4, z4, z8, 0);
#else
    __builtin_amdgcn_tensor_load_to_lds(g0, g1, z4, z4, 0);
#endif
}
#else
#define HAVE_TDM 0
#endif

// Fallback cooperative staging (f16 -> padded LDS), 256 threads.
__device__ __forceinline__ void stage_tile_fallback(const _Float16* gsrc, _Float16* lds,
                                                    int rows)
{
    const int tid = threadIdx.x;
    const int total = rows * (KSTEP / 8);               // uint4 chunks
    for (int i = tid; i < total; i += 256) {
        const int row = i / (KSTEP / 8);
        const int c8  = (i % (KSTEP / 8)) * 8;
        *(uint4*)(lds + row * LDS_P + c8) =
            *(const uint4*)(gsrc + (size_t)row * DIM + c8);
    }
}

// ======================================================================
// GEMM core: acc(16x64 per wave) += A[m0+m, k] * W[n0+n, k], K = DIM.
// A,W f16 row-major [*][DIM]. TDM double-buffered staging.
// ======================================================================
__device__ __forceinline__ void gemm_core(const _Float16* __restrict__ A,
                                          const _Float16* __restrict__ W,
                                          int m0, int n0, int a_rows,
                                          _Float16* ldsA, _Float16* ldsW,
                                          v8f acc[4])
{
    const int tid  = threadIdx.x;
    const int lane = tid & 31;
    const int wave = tid >> 5;
    const int g    = lane >> 4;
    const int l15  = lane & 15;
    const int NSTEP = DIM / KSTEP;                      // 16

#if HAVE_TDM
    const unsigned ldsA_b = (unsigned)(unsigned long long)(void*)ldsA;
    const unsigned ldsW_b = (unsigned)(unsigned long long)(void*)ldsW;
    if (wave == 0) {
        tdm_load_tile(A + (size_t)m0 * DIM, ldsA_b, a_rows, 128);
        tdm_load_tile(W + (size_t)n0 * DIM, ldsW_b, DIM, 64);
    }
#endif

    for (int ks = 0; ks < NSTEP; ++ks) {
        const int cur = ks & 1;
#if HAVE_TDM
        if (wave == 0) {
            if (ks + 1 < NSTEP) {
                const int k1 = (ks + 1) * KSTEP;
                tdm_load_tile(A + (size_t)m0 * DIM + k1,
                              ldsA_b + (unsigned)((cur ^ 1) * ABUF_H * 2), a_rows, 128);
                tdm_load_tile(W + (size_t)n0 * DIM + k1,
                              ldsW_b + (unsigned)((cur ^ 1) * WBUF_H * 2), DIM, 64);
                __builtin_amdgcn_s_wait_tensorcnt(2);   // current pair retired
            } else {
                __builtin_amdgcn_s_wait_tensorcnt(0);
            }
        }
        __syncthreads();                                // publish current buffer
        const _Float16* bufA = ldsA + cur * ABUF_H;
        const _Float16* bufW = ldsW + cur * WBUF_H;
#else
        const int k0 = ks * KSTEP;
        stage_tile_fallback(A + (size_t)m0 * DIM + k0, ldsA, 128);
        stage_tile_fallback(W + (size_t)n0 * DIM + k0, ldsW, 64);
        __syncthreads();
        const _Float16* bufA = ldsA;
        const _Float16* bufW = ldsW;
        (void)cur;
#endif
        #pragma unroll
        for (int c = 0; c < 2; ++c) {                   // two K=32 chunks
            Frag16 a;
            const _Float16* ar = bufA + (wave * 16 + l15) * LDS_P + 32 * c;
            a.q[0] = *(const uint4*)(ar + 8 * g);
            a.q[1] = *(const uint4*)(ar + 16 + 8 * g);
            #pragma unroll
            for (int nt = 0; nt < 4; ++nt) {
                Frag16 b;
                const _Float16* br = bufW + (nt * 16 + l15) * LDS_P + 32 * c;
                b.q[0] = *(const uint4*)(br + 8 * g);
                b.q[1] = *(const uint4*)(br + 16 + 8 * g);
                acc[nt] = wmma_f16(a, b, acc[nt]);
            }
        }
        __syncthreads();                                // done reading buffer
    }
}

// ======================================================================
// Kernel 0: f32 -> f16 convert (x and the four weight matrices).
// ======================================================================
__global__ void __launch_bounds__(256)
cvt_f16_kernel(const float* __restrict__ src, _Float16* __restrict__ dst, int n4)
{
    const int i = blockIdx.x * 256 + threadIdx.x;
    if (i < n4) {
        const float4 f = ((const float4*)src)[i];
        union { _Float16 h[4]; uint2 u; } o;
        o.h[0] = (_Float16)f.x; o.h[1] = (_Float16)f.y;
        o.h[2] = (_Float16)f.z; o.h[3] = (_Float16)f.w;
        ((uint2*)dst)[i] = o.u;
    }
}

// ======================================================================
// Kernel 1: QKV projection. grid = (32, 16, 3), 256 threads.
//   z=0: Q -> [n][h][s][hd]   z=1: K -> [n][h][s][hd]
//   z=2: V -> [n][h][hd][s]  (transposed for the PV GEMM)
// ======================================================================
__global__ void __launch_bounds__(256)
proj_qkv_kernel(const _Float16* __restrict__ xh,
                const _Float16* __restrict__ Wqh, const float* __restrict__ bq,
                const _Float16* __restrict__ Wkh, const float* __restrict__ bk,
                const _Float16* __restrict__ Wvh, const float* __restrict__ bv,
                _Float16* __restrict__ Qh, _Float16* __restrict__ Kh,
                _Float16* __restrict__ Vt)
{
    __shared__ __align__(16) _Float16 ldsA[2 * ABUF_H];
    __shared__ __align__(16) _Float16 ldsW[2 * WBUF_H];

    const int m0 = blockIdx.x * 128;
    const int n0 = blockIdx.y * 64;
    const int z  = blockIdx.z;
    const _Float16* W    = (z == 0) ? Wqh : (z == 1) ? Wkh : Wvh;
    const float*    bias = (z == 0) ? bq  : (z == 1) ? bk  : bv;

    v8f acc[4] = {};
    gemm_core(xh, W, m0, n0, MROWS, ldsA, ldsW, acc);

    const int lane = threadIdx.x & 31;
    const int wave = threadIdx.x >> 5;
    const int g = lane >> 4, l15 = lane & 15;

    #pragma unroll
    for (int nt = 0; nt < 4; ++nt) {
        const int col = n0 + nt * 16 + l15;
        const float bv_ = bias[col];
        const int h  = col >> 6;
        const int hd = col & 63;
        #pragma unroll
        for (int r = 0; r < 8; ++r) {
            const int m  = m0 + wave * 16 + r + 8 * g;
            const int nb = m >> 11;
            const int s  = m & (SEQ - 1);
            const _Float16 hv = (_Float16)(acc[nt][r] + bv_);
            const size_t head = (size_t)(nb * NHEADS + h);
            if (z == 0)      Qh[(head * SEQ + s) * HD + hd] = hv;
            else if (z == 1) Kh[(head * SEQ + s) * HD + hd] = hv;
            else             Vt[(head * HD + hd) * SEQ + s] = hv;
        }
    }
}

// ======================================================================
// Kernel 2: attention. grid = (S/16, N*H), 128 threads = 4 waves.
// Whole 16x2048 score block resident in LDS (~197 KB of 320 KB WGP LDS).
// ======================================================================
__global__ void __launch_bounds__(128)
attn_kernel(const _Float16* __restrict__ Qh, const _Float16* __restrict__ Kh,
            const _Float16* __restrict__ Vt, const int* __restrict__ mask,
            _Float16* __restrict__ O)
{
    extern __shared__ __align__(16) unsigned char smem[];
    float*    scores  = (float*)smem;                               // 16*2048 f32
    _Float16* attn    = (_Float16*)(smem + 16 * SEQ * 4);           // 16*2048 f16
    float*    rowsum  = (float*)(smem + 16 * SEQ * 4 + 16 * SEQ * 2);
    int*      rowmask = (int*)(rowsum + 16);

    const int tid  = threadIdx.x;
    const int lane = tid & 31;
    const int wave = tid >> 5;
    const int g = lane >> 4, l15 = lane & 15;

    const int nh = blockIdx.y;
    const int nb = nh >> 4, h = nh & 15;
    const int s0 = blockIdx.x * 16;

    if (tid < 16) rowmask[tid] = mask[nb * SEQ + s0 + tid];
    __syncthreads();

    const size_t headQK = (size_t)(nb * NHEADS + h) * SEQ * HD;

    // Q A-fragments (two Hd-chunks of 32), reused across all k-tiles
    Frag16 aq0, aq1;
    {
        const _Float16* qr = Qh + headQK + (size_t)(s0 + l15) * HD;
        aq0.q[0] = *(const uint4*)(qr + 8 * g);
        aq0.q[1] = *(const uint4*)(qr + 16 + 8 * g);
        aq1.q[0] = *(const uint4*)(qr + 32 + 8 * g);
        aq1.q[1] = *(const uint4*)(qr + 48 + 8 * g);
    }

    // scores = (Q K^T) / sqrt(Hd), masked, into LDS
    for (int kt = wave; kt < SEQ / 16; kt += 4) {
        Frag16 b0, b1;
        const _Float16* kr = Kh + headQK + (size_t)(kt * 16 + l15) * HD;
        b0.q[0] = *(const uint4*)(kr + 8 * g);
        b0.q[1] = *(const uint4*)(kr + 16 + 8 * g);
        b1.q[0] = *(const uint4*)(kr + 32 + 8 * g);
        b1.q[1] = *(const uint4*)(kr + 48 + 8 * g);
        v8f c = {};
        c = wmma_f16(aq0, b0, c);
        c = wmma_f16(aq1, b1, c);
        #pragma unroll
        for (int r = 0; r < 8; ++r) {
            const int row = r + 8 * g;
            const float v = rowmask[row] ? (c[r] * 0.125f) : MASKV;
            scores[row * SEQ + kt * 16 + l15] = v;
        }
    }
    __syncthreads();

    // row softmax (wave32 shuffle reductions), keep unnormalized exp in f16
    for (int row = wave; row < 16; row += 4) {
        const float* sr = scores + row * SEQ;
        float mx = -3.0e38f;
        for (int c = lane; c < SEQ; c += 32) mx = fmaxf(mx, sr[c]);
        #pragma unroll
        for (int off = 16; off; off >>= 1) mx = fmaxf(mx, __shfl_xor(mx, off));
        float sum = 0.0f;
        _Float16* ar = attn + row * SEQ;
        for (int c = lane; c < SEQ; c += 32) {
            const float e = __expf(sr[c] - mx);
            sum += e;
            ar[c] = (_Float16)e;
        }
        #pragma unroll
        for (int off = 16; off; off >>= 1) sum += __shfl_xor(sum, off);
        if (lane == 0) rowsum[row] = sum;
    }
    __syncthreads();

    // out(16x64) = attn(16xS) @ V(SxHd); wave owns one 16-wide d-tile.
    // Manually double-buffered: fragments for step k+1 load before WMMA k.
    const int d0 = wave * 16;
    const _Float16* vrow = Vt + (size_t)(nb * NHEADS + h) * HD * SEQ
                              + (size_t)(d0 + l15) * SEQ;
    v8f c = {};
    Frag16 af, bf;
    {
        const _Float16* ar = attn + l15 * SEQ;
        af.q[0] = *(const uint4*)(ar + 8 * g);
        af.q[1] = *(const uint4*)(ar + 16 + 8 * g);
        bf.q[0] = *(const uint4*)(vrow + 8 * g);
        bf.q[1] = *(const uint4*)(vrow + 16 + 8 * g);
    }
    for (int kc = 0; kc < SEQ; kc += 32) {
        Frag16 afn, bfn;
        if (kc + 32 < SEQ) {
            const _Float16* ar = attn + l15 * SEQ + kc + 32;
            afn.q[0] = *(const uint4*)(ar + 8 * g);
            afn.q[1] = *(const uint4*)(ar + 16 + 8 * g);
            const _Float16* vr = vrow + kc + 32;
            bfn.q[0] = *(const uint4*)(vr + 8 * g);
            bfn.q[1] = *(const uint4*)(vr + 16 + 8 * g);
        }
        c = wmma_f16(af, bf, c);
        if (kc + 32 < SEQ) { af = afn; bf = bfn; }
    }
    #pragma unroll
    for (int r = 0; r < 8; ++r) {
        const int row = r + 8 * g;
        const float val = c[r] / rowsum[row];
        O[((size_t)(nb * SEQ + s0 + row)) * DIM + h * HD + d0 + l15] = (_Float16)val;
    }
}

// ======================================================================
// Kernel 3: output projection (A = f16 attention output, f32 result).
// ======================================================================
__global__ void __launch_bounds__(256)
proj_out_kernel(const _Float16* __restrict__ O, const _Float16* __restrict__ Woh,
                const float* __restrict__ bo, float* __restrict__ out)
{
    __shared__ __align__(16) _Float16 ldsA[2 * ABUF_H];
    __shared__ __align__(16) _Float16 ldsW[2 * WBUF_H];

    const int m0 = blockIdx.x * 128;
    const int n0 = blockIdx.y * 64;

    v8f acc[4] = {};
    gemm_core(O, Woh, m0, n0, MROWS, ldsA, ldsW, acc);

    const int lane = threadIdx.x & 31;
    const int wave = threadIdx.x >> 5;
    const int g = lane >> 4, l15 = lane & 15;

    #pragma unroll
    for (int nt = 0; nt < 4; ++nt) {
        const int col = n0 + nt * 16 + l15;
        const float bv_ = bo[col];
        #pragma unroll
        for (int r = 0; r < 8; ++r) {
            const int m = m0 + wave * 16 + r + 8 * g;
            out[(size_t)m * DIM + col] = acc[nt][r] + bv_;
        }
    }
}

// ======================================================================
extern "C" void kernel_launch(void* const* d_in, const int* in_sizes, int n_in,
                              void* d_out, int out_size, void* d_ws, size_t ws_size,
                              hipStream_t stream)
{
    const float* x    = (const float*)d_in[0];
    const int*   mask = (const int*)  d_in[1];
    const float* Wq   = (const float*)d_in[2];
    const float* bq   = (const float*)d_in[3];
    const float* Wk   = (const float*)d_in[4];
    const float* bk   = (const float*)d_in[5];
    const float* Wv   = (const float*)d_in[6];
    const float* bv   = (const float*)d_in[7];
    const float* Wo   = (const float*)d_in[8];
    const float* bo   = (const float*)d_in[9];
    float* out = (float*)d_out;

    const size_t xN   = (size_t)MROWS * DIM;     // 4M elems
    const size_t wN   = (size_t)DIM * DIM;       // 1M elems
    _Float16* xh  = (_Float16*)d_ws;             // 4M halves
    _Float16* Wqh = xh  + xN;
    _Float16* Wkh = Wqh + wN;
    _Float16* Wvh = Wkh + wN;
    _Float16* Woh = Wvh + wN;
    _Float16* Qh  = Woh + wN;                    // per-matrix: 4M halves
    _Float16* Kh  = Qh + xN;
    _Float16* Vt  = Kh + xN;
    _Float16* O   = Vt + xN;                     // total ws: 48 MB f16

    // f32 -> f16 pre-conversion (makes every GEMM operand TDM-stageable)
    cvt_f16_kernel<<<(int)(xN / 4 + 255) / 256, 256, 0, stream>>>(x,  xh,  (int)(xN / 4));
    cvt_f16_kernel<<<(int)(wN / 4 + 255) / 256, 256, 0, stream>>>(Wq, Wqh, (int)(wN / 4));
    cvt_f16_kernel<<<(int)(wN / 4 + 255) / 256, 256, 0, stream>>>(Wk, Wkh, (int)(wN / 4));
    cvt_f16_kernel<<<(int)(wN / 4 + 255) / 256, 256, 0, stream>>>(Wv, Wvh, (int)(wN / 4));
    cvt_f16_kernel<<<(int)(wN / 4 + 255) / 256, 256, 0, stream>>>(Wo, Woh, (int)(wN / 4));

    proj_qkv_kernel<<<dim3(MROWS / 128, DIM / 64, 3), 256, 0, stream>>>(
        xh, Wqh, bq, Wkh, bk, Wvh, bv, Qh, Kh, Vt);

    const size_t attnLds = (size_t)16 * SEQ * 4 + (size_t)16 * SEQ * 2
                         + 16 * sizeof(float) + 16 * sizeof(int);
    attn_kernel<<<dim3(SEQ / 16, NBATCH * NHEADS), 128, attnLds, stream>>>(
        Qh, Kh, Vt, mask, O);

    proj_out_kernel<<<dim3(MROWS / 128, DIM / 64), 256, 0, stream>>>(
        O, Woh, bo, out);
}
